// Attention_386547056666
// MI455X (gfx1250) — compile-verified, matrix operand order checked by reading, not run on hardware
//
#include <hip/hip_runtime.h>
#include <hip/hip_bf16.h>

typedef _Float16 f16;
typedef _Float16 v16h __attribute__((ext_vector_type(16)));
typedef _Float16 v8h  __attribute__((ext_vector_type(8)));
typedef float    v8f  __attribute__((ext_vector_type(8)));

#define B_SZ   16
#define NSEQ   1024
#define HEADS  8
#define DIMH   32
#define DINP   256
#define NQKV   768
#define NOUT   256
#define SCALE  0.17677669529663689f   // 32^-0.5

static __device__ __forceinline__ v16h cat8(v8h lo, v8h hi) {
  return __builtin_shufflevector(lo, hi, 0,1,2,3,4,5,6,7,8,9,10,11,12,13,14,15);
}
static __device__ __forceinline__ v8f wmma16(v16h a, v16h b, v8f c) {
  // D = A(16x32 f16) * B(32x16 f16) + C(16x16 f32)
  return __builtin_amdgcn_wmma_f32_16x16x32_f16(false, a, false, b, (short)0, c, false, false);
}

// LDS byte offset of a shared-memory pointer: low 32 bits of the flat address
// (see ISA 10.2 on the LDS aperture).
static __device__ __forceinline__ unsigned lds_u32(const void* p) {
  return (unsigned)(size_t)p;
}
// Async copy of 16 bytes global -> LDS per lane (ASYNCcnt-tracked).
static __device__ __forceinline__ void async_cp16(unsigned lds_dst, const f16* gsrc) {
  asm volatile("global_load_async_to_lds_b128 %0, %1, off"
               :: "v"(lds_dst), "v"(gsrc)
               : "memory");
}
#define WAIT_ASYNC_0() asm volatile("s_wait_asynccnt 0x0" ::: "memory")
#define WAIT_ASYNC_1() asm volatile("s_wait_asynccnt 0x1" ::: "memory")

// ---------------------------------------------------------------------------
// Kernel 1: transpose weights to f16 [out, k] so WMMA B fragments are
// contiguous per-lane 16B loads.
// ---------------------------------------------------------------------------
__global__ void __launch_bounds__(256) k_prep(const float* __restrict__ Wqkv,
                                              const float* __restrict__ Wout,
                                              f16* __restrict__ WqkvT,
                                              f16* __restrict__ WoutT) {
  int idx = blockIdx.x * 256 + threadIdx.x;           // 1024 blocks -> 262144
  if (idx < NQKV * DINP) {
    int o = idx / DINP, k = idx % DINP;
    WqkvT[o * DINP + k] = (f16)Wqkv[k * NQKV + o];
  } else {
    int j = idx - NQKV * DINP;                        // < 256*256
    int o = j / DINP, k = j % DINP;
    WoutT[o * DINP + k] = (f16)Wout[k * NOUT + o];
  }
}

// ---------------------------------------------------------------------------
// Kernel 2: materialize relative-position bias [H][1024][1024] as f16.
// ---------------------------------------------------------------------------
__global__ void __launch_bounds__(256) k_bias(const float* __restrict__ table,
                                              const int* __restrict__ relidx,
                                              f16* __restrict__ biasF) {
  unsigned idx = blockIdx.x * 256u + threadIdx.x;     // 32768 blocks -> 8M
  unsigned h = idx >> 20;
  unsigned r = idx & 0xFFFFFu;
  biasF[idx] = (f16)table[relidx[r] * HEADS + h];
}

// ---------------------------------------------------------------------------
// Kernel 3: QKV projection. x[16384,256]f32 @ WqkvT -> q/k/vT in head layout.
// Block = 8 waves stacked in M (128x64 C tile). The shared 64x256 f16 weight
// panel is staged once into LDS via async-to-LDS copies (8x traffic cut).
// ---------------------------------------------------------------------------
__global__ void __launch_bounds__(256) k_qkv(const float* __restrict__ x,
                                             const f16* __restrict__ WT,
                                             f16* __restrict__ qw,
                                             f16* __restrict__ kw,
                                             f16* __restrict__ vT) {
  __shared__ __align__(16) f16 Bpanel[64][DINP];      // 32 KB
  const int tid  = threadIdx.x;
  const int lane = tid & 31;
  const int w    = tid >> 5;
  const int lrow = lane & 15;
  const int lhi  = lane >> 4;
  const int m0   = blockIdx.y * 128 + w * 16;
  const int n0   = blockIdx.x * 64;

  // Stage weight panel: 64 rows x 512 B = 2048 chunks of 16 B, 8 per thread.
  #pragma unroll
  for (int j = 0; j < 8; ++j) {
    const int c = tid + j * 256;                      // chunk id
    const int row = c >> 5, col8 = (c & 31) * 8;
    async_cp16(lds_u32(&Bpanel[row][col8]),
               WT + (size_t)(n0 + row) * DINP + col8);
  }
  WAIT_ASYNC_0();
  __syncthreads();

  v8f acc[4];
  #pragma unroll
  for (int t = 0; t < 4; ++t) acc[t] = (v8f){0.f,0.f,0.f,0.f,0.f,0.f,0.f,0.f};

  const float* xrow = x + (size_t)(m0 + lrow) * DINP;
  #pragma unroll
  for (int k0 = 0; k0 < DINP; k0 += 32) {
    // A fragment: f32 -> f16 inline conversion (K chunks lhi*8 and 16+lhi*8)
    const float* xp = xrow + k0 + lhi * 8;
    v8h alo, ahi;
    #pragma unroll
    for (int i = 0; i < 8; ++i) { alo[i] = (f16)xp[i]; ahi[i] = (f16)xp[16 + i]; }
    v16h a = cat8(alo, ahi);
    #pragma unroll
    for (int t = 0; t < 4; ++t) {
      const f16* bp = &Bpanel[t * 16 + lrow][k0 + lhi * 8];
      v16h b = cat8(*(const v8h*)bp, *(const v8h*)(bp + 16));
      acc[t] = wmma16(a, b, acc[t]);
    }
  }

  // Scatter: C element (row = r + 8*lhi, col = lrow) per the 16x16 f32 layout.
  #pragma unroll
  for (int t = 0; t < 4; ++t) {
    const int gcol  = n0 + t * 16 + lrow;
    const int which = gcol >> 8;        // 0=q 1=k 2=v
    const int rem   = gcol & 255;
    const int h     = rem >> 5;
    const int d     = rem & 31;
    #pragma unroll
    for (int r = 0; r < 8; ++r) {
      const int grow = m0 + r + 8 * lhi;
      const int bb   = grow >> 10;
      const int n    = grow & 1023;
      const float v  = acc[t][r];
      const size_t hd = ((size_t)(bb * HEADS + h) * NSEQ + n) * DIMH + d;
      if (which == 0)      qw[hd] = (f16)(v * SCALE);
      else if (which == 1) kw[hd] = (f16)v;
      else                 vT[((size_t)(bb * HEADS + h) * DIMH + d) * NSEQ + n] = (f16)v;
    }
  }
}

// ---------------------------------------------------------------------------
// Kernel 4: flash attention. One wave owns 16 query rows; all 8 waves of a
// block share one (b,h), so the 32-key K/V tiles are staged cooperatively
// into double-buffered LDS with async-to-LDS copies overlapping compute.
// Per 32-key block: 2 WMMAs (QK^T) + online softmax + 2 WMMAs (PV).
// ---------------------------------------------------------------------------
__global__ void __launch_bounds__(256) k_attn(const f16* __restrict__ qw,
                                              const f16* __restrict__ kw,
                                              const f16* __restrict__ vT,
                                              const f16* __restrict__ biasF,
                                              f16* __restrict__ attn) {
  __shared__ __align__(16) f16 Kbuf[2][32 * 32];      // [key][d]   2 x 2 KB
  __shared__ __align__(16) f16 Vbuf[2][32 * 32];      // [d][key]   2 x 2 KB
  __shared__ __align__(16) f16 Pbuf[8][512];          // per-wave P transpose
  const int tid  = threadIdx.x;
  const int lane = tid & 31;
  const int w    = tid >> 5;
  const int lrow = lane & 15;
  const int lhi  = lane >> 4;
  const int q0   = blockIdx.x * 128 + w * 16;
  const int h    = blockIdx.y;
  const int b    = blockIdx.z;
  const size_t headoff = (size_t)(b * HEADS + h) * NSEQ * DIMH;

  // Copier role: threads 0-127 move the K tile, 128-255 the V tile.
  const int cr = (tid & 127) >> 2;                    // row within 32x32 tile
  const int cc = (tid & 3) * 8;                       // half-offset in row
  const f16* csrc0;
  size_t cstep;                                       // halves per key-block
  unsigned cdst[2];
  if (tid < 128) {                                    // K: row=key, 32 halves
    csrc0 = kw + headoff + (size_t)cr * DIMH + cc;
    cstep = 32 * DIMH;
    cdst[0] = lds_u32(&Kbuf[0][cr * 32 + cc]);
    cdst[1] = lds_u32(&Kbuf[1][cr * 32 + cc]);
  } else {                                            // V: row=d, stride NSEQ
    csrc0 = vT + headoff + (size_t)cr * NSEQ + cc;
    cstep = 32;
    cdst[0] = lds_u32(&Vbuf[0][cr * 32 + cc]);
    cdst[1] = lds_u32(&Vbuf[1][cr * 32 + cc]);
  }

  // Q A-fragment (16 rows x K=32), pre-scaled by 1/sqrt(d) in k_qkv.
  const f16* qp = qw + headoff + (size_t)(q0 + lrow) * DIMH + lhi * 8;
  const v16h qa = cat8(*(const v8h*)qp, *(const v8h*)(qp + 16));

  const f16* bh = biasF + (size_t)h * NSEQ * NSEQ;

  float m[8], l[8];
  #pragma unroll
  for (int r = 0; r < 8; ++r) { m[r] = -1e30f; l[r] = 0.f; }
  const v8f zero = (v8f){0.f,0.f,0.f,0.f,0.f,0.f,0.f,0.f};
  v8f o0 = zero, o1 = zero;

  async_cp16(cdst[0], csrc0);                         // prefetch tile 0

  for (int it = 0; it < 32; ++it) {
    const int cur = it & 1;
    const int kb  = it * 32;
    __syncthreads();                  // buf[1-cur] free (prev compute done)
    if (it < 31) {
      async_cp16(cdst[1 - cur], csrc0 + (size_t)(it + 1) * cstep);
      WAIT_ASYNC_1();                 // tile `it` landed, tile it+1 in flight
    } else {
      WAIT_ASYNC_0();
    }
    __syncthreads();                  // tile `it` visible block-wide

    // K^T B-fragments from LDS: lane col = key, K-dim = d.
    const f16* kp0 = &Kbuf[cur][lrow * 32 + lhi * 8];
    const f16* kp1 = &Kbuf[cur][(16 + lrow) * 32 + lhi * 8];
    v16h bk0 = cat8(*(const v8h*)kp0, *(const v8h*)(kp0 + 16));
    v16h bk1 = cat8(*(const v8h*)kp1, *(const v8h*)(kp1 + 16));
    v8f s0 = wmma16(qa, bk0, zero);
    v8f s1 = wmma16(qa, bk1, zero);

    // Relative position bias.
    #pragma unroll
    for (int r = 0; r < 8; ++r) {
      const size_t qi = (size_t)(q0 + r + 8 * lhi) * NSEQ + kb;
      s0[r] += (float)bh[qi + lrow];
      s1[r] += (float)bh[qi + 16 + lrow];
    }

    // Online softmax: each row's 16 cols live in one 16-lane group.
    #pragma unroll
    for (int r = 0; r < 8; ++r) {
      float c = fmaxf(s0[r], s1[r]);
      c = fmaxf(c, __shfl_xor(c, 1));
      c = fmaxf(c, __shfl_xor(c, 2));
      c = fmaxf(c, __shfl_xor(c, 4));
      c = fmaxf(c, __shfl_xor(c, 8));
      const float mn   = fmaxf(m[r], c);
      const float corr = __expf(m[r] - mn);
      const float p0   = __expf(s0[r] - mn);
      const float p1   = __expf(s1[r] - mn);
      float rs = p0 + p1;
      rs += __shfl_xor(rs, 1);
      rs += __shfl_xor(rs, 2);
      rs += __shfl_xor(rs, 4);
      rs += __shfl_xor(rs, 8);
      l[r] = l[r] * corr + rs;
      m[r] = mn;
      o0[r] *= corr;
      o1[r] *= corr;
      s0[r] = p0;
      s1[r] = p1;
    }

    // Transpose P from C layout to A layout through per-wave LDS region
    // (LDS ops are in-order within a wave).
    #pragma unroll
    for (int r = 0; r < 8; ++r) {
      const int prow = r + 8 * lhi;
      Pbuf[w][prow * 32 + lrow]      = (f16)s0[r];
      Pbuf[w][prow * 32 + 16 + lrow] = (f16)s1[r];
    }
    __builtin_amdgcn_wave_barrier();
    const f16* pp = &Pbuf[w][lrow * 32 + lhi * 8];
    v16h pa = cat8(*(const v8h*)pp, *(const v8h*)(pp + 16));

    // V B-fragments from LDS: lane col = d, K-dim = key.
    const f16* vp0 = &Vbuf[cur][lrow * 32 + lhi * 8];
    const f16* vp1 = &Vbuf[cur][(16 + lrow) * 32 + lhi * 8];
    v16h bv0 = cat8(*(const v8h*)vp0, *(const v8h*)(vp0 + 16));
    v16h bv1 = cat8(*(const v8h*)vp1, *(const v8h*)(vp1 + 16));
    o0 = wmma16(pa, bv0, o0);
    o1 = wmma16(pa, bv1, o1);
  }

  // Normalize and store to [B, N, H*D] f16 for the output GEMM.
  #pragma unroll
  for (int r = 0; r < 8; ++r) {
    const float inv = 1.f / l[r];
    const int row = q0 + r + 8 * lhi;
    f16* op = attn + ((size_t)(b * NSEQ + row) * HEADS + h) * DIMH;
    op[lrow]      = (f16)(o0[r] * inv);
    op[16 + lrow] = (f16)(o1[r] * inv);
  }
}

// ---------------------------------------------------------------------------
// Kernel 5: output projection. attn[16384,256]f16 @ WoutT + b_out -> f32 out.
// Same LDS-staged weight panel as k_qkv.
// ---------------------------------------------------------------------------
__global__ void __launch_bounds__(256) k_out(const f16* __restrict__ A,
                                             const f16* __restrict__ WT,
                                             const float* __restrict__ bias,
                                             float* __restrict__ out) {
  __shared__ __align__(16) f16 Bpanel[64][DINP];      // 32 KB
  const int tid  = threadIdx.x;
  const int lane = tid & 31;
  const int w    = tid >> 5;
  const int lrow = lane & 15;
  const int lhi  = lane >> 4;
  const int m0   = blockIdx.y * 128 + w * 16;
  const int n0   = blockIdx.x * 64;

  #pragma unroll
  for (int j = 0; j < 8; ++j) {
    const int c = tid + j * 256;
    const int row = c >> 5, col8 = (c & 31) * 8;
    async_cp16(lds_u32(&Bpanel[row][col8]),
               WT + (size_t)(n0 + row) * DINP + col8);
  }
  WAIT_ASYNC_0();
  __syncthreads();

  v8f acc[4];
  #pragma unroll
  for (int t = 0; t < 4; ++t) acc[t] = (v8f){0.f,0.f,0.f,0.f,0.f,0.f,0.f,0.f};

  const f16* arow = A + (size_t)(m0 + lrow) * DINP;
  #pragma unroll
  for (int k0 = 0; k0 < DINP; k0 += 32) {
    const f16* ap = arow + k0 + lhi * 8;
    v16h a = cat8(*(const v8h*)ap, *(const v8h*)(ap + 16));
    #pragma unroll
    for (int t = 0; t < 4; ++t) {
      const f16* bp = &Bpanel[t * 16 + lrow][k0 + lhi * 8];
      v16h b = cat8(*(const v8h*)bp, *(const v8h*)(bp + 16));
      acc[t] = wmma16(a, b, acc[t]);
    }
  }

  #pragma unroll
  for (int t = 0; t < 4; ++t) {
    const int gcol = n0 + t * 16 + lrow;
    const float bv = bias[gcol];
    #pragma unroll
    for (int r = 0; r < 8; ++r) {
      out[(size_t)(m0 + r + 8 * lhi) * NOUT + gcol] = acc[t][r] + bv;
    }
  }
}

// ---------------------------------------------------------------------------
extern "C" void kernel_launch(void* const* d_in, const int* in_sizes, int n_in,
                              void* d_out, int out_size, void* d_ws, size_t ws_size,
                              hipStream_t stream) {
  const float* x      = (const float*)d_in[0];   // [16,1024,256]
  const float* W_qkv  = (const float*)d_in[1];   // [256,768]
  const float* W_out  = (const float*)d_in[2];   // [256,256]
  const float* b_out  = (const float*)d_in[3];   // [256]
  const float* btable = (const float*)d_in[4];   // [3969,8]
  const int*   relidx = (const int*)d_in[5];     // [1024*1024]
  float* out = (float*)d_out;

  char* ws = (char*)d_ws;
  size_t off = 0;
  f16* WqkvT = (f16*)(ws + off); off += (size_t)NQKV * DINP * 2;          // 384 KB
  f16* WoutT = (f16*)(ws + off); off += (size_t)NOUT * DINP * 2;          // 128 KB
  f16* qw    = (f16*)(ws + off); off += (size_t)B_SZ * HEADS * NSEQ * DIMH * 2; // 8 MB
  f16* kw    = (f16*)(ws + off); off += (size_t)B_SZ * HEADS * NSEQ * DIMH * 2; // 8 MB
  f16* vT    = (f16*)(ws + off); off += (size_t)B_SZ * HEADS * NSEQ * DIMH * 2; // 8 MB
  f16* attn  = (f16*)(ws + off); off += (size_t)B_SZ * NSEQ * HEADS * DIMH * 2; // 8 MB
  f16* biasF = (f16*)(ws + off); off += (size_t)HEADS * NSEQ * NSEQ * 2;        // 16 MB

  k_prep<<<(NQKV * DINP + NOUT * DINP) / 256, 256, 0, stream>>>(W_qkv, W_out, WqkvT, WoutT);
  k_bias<<<(HEADS * NSEQ * NSEQ) / 256, 256, 0, stream>>>(btable, relidx, biasF);
  k_qkv<<<dim3(NQKV / 64, (B_SZ * NSEQ) / 128), 256, 0, stream>>>(x, WqkvT, qw, kw, vT);
  k_attn<<<dim3(NSEQ / 128, HEADS, B_SZ), 256, 0, stream>>>(qw, kw, vT, biasF, attn);
  k_out<<<dim3(NOUT / 64, (B_SZ * NSEQ) / 128), 256, 0, stream>>>(attn, WoutT, b_out, out);
}
